// VisionTransformer_44942537785755
// MI455X (gfx1250) — matrix operand order repaired and checked by reading, not compiled, not run on hardware
//
#include <hip/hip_runtime.h>
#include <cstdint>

// ---------------------------------------------------------------------------
// ViT-Base forward on MI455X (gfx1250): bf16 WMMA everywhere.
// GEMM: 128x128x32 block tile, 8 waves, each wave 32x64 = 8 WMMAs per K-step.
// A-tile staged with GLOBAL_LOAD_ASYNC_TO_LDS_B128 (ASYNCcnt), B-tile staged
// transposed with vector loads + LDS scatter; global_prefetch_b8 on B.
// ---------------------------------------------------------------------------

typedef __attribute__((ext_vector_type(16))) __bf16 v16bf;
typedef __attribute__((ext_vector_type(8)))  float  v8f;

constexpr int BATCH  = 32;
constexpr int CIN    = 3;
constexpr int IMGSZ  = 224;
constexpr int PATCH  = 16;
constexpr int EMB    = 768;
constexpr int DEP    = 12;
constexpr int NH     = 12;
constexpr int MLPD   = 3072;
constexpr int GRIDP  = 14;               // IMG / P
constexpr int NPATCH = GRIDP * GRIDP;    // 196
constexpr int SEQ    = NPATCH + 1;       // 197
constexpr int HDIM   = EMB / NH;         // 64
constexpr int SPAD   = 224;              // SEQ padded to multiple of 32
constexpr int NCLS   = 10;
constexpr int MROWS  = BATCH * SEQ;      // 6304
constexpr int MPATCH = BATCH * NPATCH;   // 6272
constexpr int BH     = BATCH * NH;       // 384

// GEMM tiling
constexpr int TM  = 128;
constexpr int TN  = 128;
constexpr int LDK = 40;   // LDS row stride in halves (80B: 16B aligned, bank-skewed)

// Epilogue modes
constexpr int OUT_F32       = 0;  // out = alpha*acc + bias    (float)
constexpr int OUT_BF16      = 1;  // out = bf16(acc + bias)
constexpr int OUT_GELU_BF16 = 2;  // out = bf16(gelu(acc + bias))
constexpr int OUT_ADD_F32   = 3;  // out += acc + bias         (float residual)

__device__ __forceinline__ unsigned short f2bf(float f) {
  unsigned u = __float_as_uint(f);
  u += 0x7FFFu + ((u >> 16) & 1u);       // round-to-nearest-even
  return (unsigned short)(u >> 16);
}

// --------------------------- elementwise kernels ---------------------------

__global__ void k_convert(const float* __restrict__ in,
                          unsigned short* __restrict__ out, long long n) {
  long long i = (long long)blockIdx.x * blockDim.x + threadIdx.x;
  if (i < n) out[i] = f2bf(in[i]);
}

// conv_w (E, C*P*P) row-major -> wpT (C*P*P, E) bf16
__global__ void k_wpT(const float* __restrict__ w, unsigned short* __restrict__ t) {
  int idx = blockIdx.x * blockDim.x + threadIdx.x;
  if (idx >= EMB * EMB) return;
  int d = idx / EMB, e = idx % EMB;
  t[idx] = f2bf(w[e * EMB + d]);
}

// x (B,C,224,224) -> patches (B*NP, 768) bf16, inner order (c, py, px)
__global__ void k_patchify(const float* __restrict__ x, unsigned short* __restrict__ p) {
  long long idx = (long long)blockIdx.x * blockDim.x + threadIdx.x;
  if (idx >= (long long)MPATCH * EMB) return;
  int m = (int)(idx / EMB), d = (int)(idx % EMB);
  int b = m / NPATCH, pp = m % NPATCH;
  int gy = pp / GRIDP, gx = pp % GRIDP;
  int c = d >> 8, r = d & 255, py = r >> 4, px = r & 15;
  p[idx] = f2bf(x[(((long long)b * CIN + c) * IMGSZ + gy * PATCH + py) * IMGSZ
                  + gx * PATCH + px]);
}

// h[b,s,e] = (s==0 ? cls[e] : tok[b,s-1,e]) + pos[s,e]
__global__ void k_assemble(const float* __restrict__ tok, const float* __restrict__ cls,
                           const float* __restrict__ pos, float* __restrict__ h) {
  long long idx = (long long)blockIdx.x * blockDim.x + threadIdx.x;
  if (idx >= (long long)MROWS * EMB) return;
  int r = (int)(idx / EMB), e = (int)(idx % EMB);
  int b = r / SEQ, s = r % SEQ;
  float v = (s == 0) ? cls[e] : tok[((long long)b * NPATCH + (s - 1)) * EMB + e];
  h[idx] = v + pos[s * EMB + e];
}

// wave-per-row layernorm over E=768, bf16 output
__global__ void k_layernorm(const float* __restrict__ x, const float* __restrict__ w,
                            const float* __restrict__ b, unsigned short* __restrict__ y,
                            int rows) {
  int wave = threadIdx.x >> 5, lane = threadIdx.x & 31;
  int row = blockIdx.x * 8 + wave;
  if (row >= rows) return;
  const float* xr = x + (long long)row * EMB;
  float v[24]; float s = 0.f;
  #pragma unroll
  for (int i = 0; i < 24; ++i) { v[i] = xr[lane + i * 32]; s += v[i]; }
  #pragma unroll
  for (int o = 16; o >= 1; o >>= 1) s += __shfl_xor(s, o);
  float mean = s * (1.f / EMB);
  float t = 0.f;
  #pragma unroll
  for (int i = 0; i < 24; ++i) { float d = v[i] - mean; t += d * d; }
  #pragma unroll
  for (int o = 16; o >= 1; o >>= 1) t += __shfl_xor(t, o);
  float rstd = rsqrtf(t * (1.f / EMB) + 1e-5f);
  unsigned short* yr = y + (long long)row * EMB;
  #pragma unroll
  for (int i = 0; i < 24; ++i) {
    int c = lane + i * 32;
    yr[c] = f2bf((v[i] - mean) * rstd * w[c] + b[c]);
  }
}

// qkv (B,S,3E) bf16 -> q (BH,S,64), kT (BH,64,SPAD), v (BH,SPAD,64); pads zeroed
__global__ void k_reorder(const unsigned short* __restrict__ qkvb,
                          unsigned short* __restrict__ q_c,
                          unsigned short* __restrict__ kT,
                          unsigned short* __restrict__ v_c) {
  long long idx = (long long)blockIdx.x * blockDim.x + threadIdx.x;
  if (idx >= (long long)BH * SPAD * HDIM) return;
  int bh = (int)(idx / (SPAD * HDIM));
  int rem = (int)(idx % (SPAD * HDIM));
  int s = rem / HDIM, d = rem % HDIM;
  int b = bh / NH, hh = bh % NH;
  unsigned short kv = 0, vv = 0;
  if (s < SEQ) {
    const unsigned short* base = qkvb + (long long)(b * SEQ + s) * (3 * EMB);
    q_c[((long long)bh * SEQ + s) * HDIM + d] = base[hh * HDIM + d];
    kv = base[EMB + hh * HDIM + d];
    vv = base[2 * EMB + hh * HDIM + d];
  }
  kT[((long long)bh * HDIM + d) * SPAD + s] = kv;
  v_c[((long long)bh * SPAD + s) * HDIM + d] = vv;
}

// wave-per-row softmax over SPAD=224 (valid cols < 197), bf16 probs out
__global__ void k_softmax(const float* __restrict__ sc,
                          unsigned short* __restrict__ pr, int rows) {
  int wave = threadIdx.x >> 5, lane = threadIdx.x & 31;
  int row = blockIdx.x * 8 + wave;
  if (row >= rows) return;
  const float* xr = sc + (long long)row * SPAD;
  float v[7]; float m = -3.4e38f;
  #pragma unroll
  for (int i = 0; i < 7; ++i) {
    int c = i * 32 + lane;
    v[i] = (c < SEQ) ? xr[c] : -3.4e38f;
    m = fmaxf(m, v[i]);
  }
  #pragma unroll
  for (int o = 16; o >= 1; o >>= 1) m = fmaxf(m, __shfl_xor(m, o));
  float s = 0.f;
  #pragma unroll
  for (int i = 0; i < 7; ++i) {
    int c = i * 32 + lane;
    float e = (c < SEQ) ? expf(v[i] - m) : 0.f;
    v[i] = e; s += e;
  }
  #pragma unroll
  for (int o = 16; o >= 1; o >>= 1) s += __shfl_xor(s, o);
  float inv = 1.f / s;
  unsigned short* yr = pr + (long long)row * SPAD;
  #pragma unroll
  for (int i = 0; i < 7; ++i) yr[i * 32 + lane] = f2bf(v[i] * inv);
}

// final LN of cls rows + 768x10 head; one wave per batch element
__global__ void k_head(const float* __restrict__ h, const float* __restrict__ lnw,
                       const float* __restrict__ lnb, const float* __restrict__ hw,
                       const float* __restrict__ hb, float* __restrict__ out) {
  int b = blockIdx.x, lane = threadIdx.x;
  const float* row = h + (long long)b * SEQ * EMB;  // s == 0 (cls)
  float v[24]; float s = 0.f;
  #pragma unroll
  for (int i = 0; i < 24; ++i) { v[i] = row[lane + i * 32]; s += v[i]; }
  #pragma unroll
  for (int o = 16; o >= 1; o >>= 1) s += __shfl_xor(s, o);
  float mean = s * (1.f / EMB);
  float t = 0.f;
  #pragma unroll
  for (int i = 0; i < 24; ++i) { float d = v[i] - mean; t += d * d; }
  #pragma unroll
  for (int o = 16; o >= 1; o >>= 1) t += __shfl_xor(t, o);
  float rstd = rsqrtf(t * (1.f / EMB) + 1e-5f);
  float nrm[24];
  #pragma unroll
  for (int i = 0; i < 24; ++i) {
    int c = lane + i * 32;
    nrm[i] = (v[i] - mean) * rstd * lnw[c] + lnb[c];
  }
  for (int c = 0; c < NCLS; ++c) {
    float p = 0.f;
    #pragma unroll
    for (int i = 0; i < 24; ++i) p += nrm[i] * hw[(lane + i * 32) * NCLS + c];
    #pragma unroll
    for (int o = 16; o >= 1; o >>= 1) p += __shfl_xor(p, o);
    if (lane == 0) out[b * NCLS + c] = p + hb[c];
  }
}

// ------------------------------ WMMA GEMM ----------------------------------
// C[z] = alpha * A[z] (MxK, bf16 rm) * B[z] (KxN, bf16 rm) + bias, batched.
// Block: 256 threads = 8 waves; tile 128(M) x 128(N) x 32(K).
// Wave grid 4(M) x 2(N): each wave 32x64 output = 2 A-frags x 4 B-frags
//  -> 8 v_wmma_f32_16x16x32_bf16 per K-step per wave.
// OOB rows/cols are *clamped* (not masked): an OOB A-row / B-col only feeds
// output rows/cols that the guarded epilogue never stores.
__launch_bounds__(256)
__global__ void k_gemm(const unsigned short* __restrict__ A, long long strideAz,
                       const unsigned short* __restrict__ Bm, long long strideBz,
                       const float* __restrict__ bias,
                       void* __restrict__ Out, int outZdiv,
                       long long outStrideOuter, long long outStrideInner,
                       int M, int N, int K, int ldo, float alpha, int mode) {
  __shared__ alignas(16) unsigned short As[TM * LDK];
  __shared__ alignas(16) unsigned short Bs[TN * LDK];

  const int tid   = threadIdx.x;
  const int z     = blockIdx.z;
  const int mTile = blockIdx.y * TM;
  const int nTile = blockIdx.x * TN;

  const unsigned short* Az = A  + (long long)z * strideAz;
  const unsigned short* Bz = Bm + (long long)z * strideBz;

  const int lane = tid & 31;
  const int wave = tid >> 5;
  const int m0   = (wave & 3) * 32;   // 0,32,64,96
  const int n0   = (wave >> 2) * 64;  // 0,64
  const int lr   = lane & 15;
  const int hi   = lane >> 4;

  v8f acc[2][4] = {};

  const int sBk = tid & 31;            // 0..31
  const int sBn = (tid >> 5) * 16;     // 0..112

  for (int kb = 0; kb < K; kb += 32) {
    __syncthreads();
    // ---- stage A tile: async DMA global -> LDS, 16B per lane (ASYNCcnt)
    #pragma unroll
    for (int it = 0; it < 2; ++it) {
      int lin  = tid + it * 256;       // 0..511
      int row  = lin >> 2;             // 0..127
      int ch   = (lin & 3) * 8;        // halves: 0,8,16,24
      int grow = mTile + row;
      if (grow >= M) grow = M - 1;     // clamp: rows >= M are never stored
      const unsigned short* gp = Az + (long long)grow * K + kb + ch;
      unsigned lds = (unsigned)(size_t)&As[row * LDK + ch];  // LDS byte addr
      asm volatile("global_load_async_to_lds_b128 %0, %1, off"
                   :: "v"(lds), "v"(gp) : "memory");
    }
    // ---- stage B tile transposed: Bs[n][k], n in [0,TN), k in [0,32)
    #pragma unroll
    for (int vv = 0; vv < 2; ++vv) {
      int gcol = nTile + sBn + vv * 8;
      if (gcol > N - 8) gcol = N - 8;  // clamp: cols >= N are never stored
      const unsigned short* gp = Bz + (long long)(kb + sBk) * N + gcol;
      uint4 val = *reinterpret_cast<const uint4*>(gp);
      if (kb + 32 < K) __builtin_prefetch(gp + (long long)32 * N, 0, 1);
      const unsigned short* hp = reinterpret_cast<const unsigned short*>(&val);
      #pragma unroll
      for (int j = 0; j < 8; ++j)
        Bs[(sBn + vv * 8 + j) * LDK + sBk] = hp[j];
    }
    // complete this wave's async LDS writes, then cross-wave barrier
    asm volatile("s_wait_asynccnt 0x0" ::: "memory");
    __syncthreads();

    // ---- gather fragments per ISA VGPR layouts
    union Frag { v16bf v; unsigned int u[8]; };
    Frag fa[2], fb[4];
    #pragma unroll
    for (int sidx = 0; sidx < 2; ++sidx) {
      int row = m0 + sidx * 16 + lr;
      #pragma unroll
      for (int t = 0; t < 8; ++t) {
        // A 16x32 bf16: K = 2*(t&3) + 16*(t>>2) + 8*hi  (pairs per VGPR)
        int k = ((t & 3) << 1) + ((t >> 2) << 4) + (hi << 3);
        fa[sidx].u[t] = *reinterpret_cast<const unsigned int*>(&As[row * LDK + k]);
      }
    }
    #pragma unroll
    for (int sidx = 0; sidx < 4; ++sidx) {
      int col = n0 + sidx * 16 + lr;
      #pragma unroll
      for (int t = 0; t < 8; ++t) {
        // B 32x16 bf16: K = 2*(t&3) + 16*(t>>2) + 8*hi  (pairs per VGPR)
        int k = ((t & 3) << 1) + ((t >> 2) << 4) + (hi << 3);
        fb[sidx].u[t] = *reinterpret_cast<const unsigned int*>(&Bs[col * LDK + k]);
      }
    }
    #pragma unroll
    for (int i = 0; i < 2; ++i)
      #pragma unroll
      for (int j = 0; j < 4; ++j)
        acc[i][j] = __builtin_amdgcn_wmma_f32_16x16x32_bf16(
            false, fa[i].v, false, fb[j].v, (short)0, acc[i][j], false, false);
  }

  // ---- epilogue: D layout row = r + 8*hi, col = lane&15
  long long outBase = (long long)(z / outZdiv) * outStrideOuter
                    + (long long)(z % outZdiv) * outStrideInner;
  #pragma unroll
  for (int i = 0; i < 2; ++i) {
    #pragma unroll
    for (int j = 0; j < 4; ++j) {
      int colg = nTile + n0 + j * 16 + lr;
      if (colg >= N) continue;
      float bv = bias ? bias[colg] : 0.f;
      #pragma unroll
      for (int r = 0; r < 8; ++r) {
        int rowg = mTile + m0 + i * 16 + r + hi * 8;
        if (rowg >= M) continue;
        float v = acc[i][j][r] * alpha + bv;
        long long oidx = outBase + (long long)rowg * ldo + colg;
        if (mode == OUT_F32) {
          ((float*)Out)[oidx] = v;
        } else if (mode == OUT_BF16) {
          ((unsigned short*)Out)[oidx] = f2bf(v);
        } else if (mode == OUT_GELU_BF16) {
          float g = 0.5f * v * (1.f + erff(v * 0.70710678118654752f));
          ((unsigned short*)Out)[oidx] = f2bf(g);
        } else {  // OUT_ADD_F32 residual
          float* p = (float*)Out + oidx;
          *p = *p + v;
        }
      }
    }
  }
}

// ------------------------------- host side ---------------------------------

extern "C" void kernel_launch(void* const* d_in, const int* in_sizes, int n_in,
                              void* d_out, int out_size, void* d_ws, size_t ws_size,
                              hipStream_t stream) {
  const float* x       = (const float*)d_in[0];
  const float* conv_w  = (const float*)d_in[1];
  const float* conv_b  = (const float*)d_in[2];
  const float* cls_tok = (const float*)d_in[3];
  const float* pos_emb = (const float*)d_in[4];
  const float* ln1_w   = (const float*)d_in[5];
  const float* ln1_b   = (const float*)d_in[6];
  const float* qkv_w   = (const float*)d_in[7];
  const float* qkv_b   = (const float*)d_in[8];
  const float* proj_w  = (const float*)d_in[9];
  const float* proj_b  = (const float*)d_in[10];
  const float* ln2_w   = (const float*)d_in[11];
  const float* ln2_b   = (const float*)d_in[12];
  const float* mlp_w1  = (const float*)d_in[13];
  const float* mlp_b1  = (const float*)d_in[14];
  const float* mlp_w2  = (const float*)d_in[15];
  const float* mlp_b2  = (const float*)d_in[16];
  const float* lnf_w   = (const float*)d_in[17];
  const float* lnf_b   = (const float*)d_in[18];
  const float* head_w  = (const float*)d_in[19];
  const float* head_b  = (const float*)d_in[20];
  float* out = (float*)d_out;

  char* ws = (char*)d_ws;
  size_t cur = 0;
  auto alloc = [&](size_t bytes) -> char* {
    char* p = ws + cur;
    cur = (cur + bytes + 255) & ~(size_t)255;
    return p;
  };

  float*          h_f      = (float*)         alloc((size_t)MROWS * EMB * 4);
  unsigned short* yb       = (unsigned short*)alloc((size_t)MROWS * EMB * 2);
  unsigned short* qkvb     = (unsigned short*)alloc((size_t)MROWS * 3 * EMB * 2);
  unsigned short* q_c      = (unsigned short*)alloc((size_t)BH * SEQ * HDIM * 2);
  unsigned short* kT       = (unsigned short*)alloc((size_t)BH * HDIM * SPAD * 2);
  unsigned short* v_c      = (unsigned short*)alloc((size_t)BH * SPAD * HDIM * 2);
  float*          scores   = (float*)         alloc((size_t)BH * SEQ * SPAD * 4);
  unsigned short* probs    = (unsigned short*)alloc((size_t)BH * SEQ * SPAD * 2);
  unsigned short* o_b      = (unsigned short*)alloc((size_t)MROWS * EMB * 2);
  unsigned short* mb       = (unsigned short*)alloc((size_t)MROWS * MLPD * 2);
  float*          tok_f    = (float*)         alloc((size_t)MPATCH * EMB * 4);
  unsigned short* patches  = (unsigned short*)alloc((size_t)MPATCH * EMB * 2);
  unsigned short* wpT_b    = (unsigned short*)alloc((size_t)EMB * EMB * 2);
  unsigned short* qkvw_b   = (unsigned short*)alloc((size_t)DEP * EMB * 3 * EMB * 2);
  unsigned short* projw_b  = (unsigned short*)alloc((size_t)DEP * EMB * EMB * 2);
  unsigned short* w1_b     = (unsigned short*)alloc((size_t)DEP * EMB * MLPD * 2);
  unsigned short* w2_b     = (unsigned short*)alloc((size_t)DEP * MLPD * EMB * 2);
  (void)ws_size; (void)in_sizes; (void)n_in; (void)out_size;

  auto cdiv = [](long long a, long long b) { return (unsigned)((a + b - 1) / b); };

  auto gemm = [&](const void* Ap, long long sAz, const void* Bp, long long sBz,
                  const float* bias, void* Op, int zdiv, long long oOuter,
                  long long oInner, int M, int N, int K, int ldo,
                  float alpha, int mode, int Z) {
    dim3 g((N + TN - 1) / TN, (M + TM - 1) / TM, Z);
    k_gemm<<<g, dim3(256), 0, stream>>>(
        (const unsigned short*)Ap, sAz, (const unsigned short*)Bp, sBz, bias,
        Op, zdiv, oOuter, oInner, M, N, K, ldo, alpha, mode);
  };

  // ---- weight conversions (deterministic, every call)
  k_convert<<<cdiv((long long)DEP * EMB * 3 * EMB, 256), 256, 0, stream>>>(
      qkv_w, qkvw_b, (long long)DEP * EMB * 3 * EMB);
  k_convert<<<cdiv((long long)DEP * EMB * EMB, 256), 256, 0, stream>>>(
      proj_w, projw_b, (long long)DEP * EMB * EMB);
  k_convert<<<cdiv((long long)DEP * EMB * MLPD, 256), 256, 0, stream>>>(
      mlp_w1, w1_b, (long long)DEP * EMB * MLPD);
  k_convert<<<cdiv((long long)DEP * MLPD * EMB, 256), 256, 0, stream>>>(
      mlp_w2, w2_b, (long long)DEP * MLPD * EMB);
  k_wpT<<<cdiv((long long)EMB * EMB, 256), 256, 0, stream>>>(conv_w, wpT_b);

  // ---- patch embed + token assembly
  k_patchify<<<cdiv((long long)MPATCH * EMB, 256), 256, 0, stream>>>(x, patches);
  gemm(patches, 0, wpT_b, 0, conv_b, tok_f, 1, 0, 0,
       MPATCH, EMB, EMB, EMB, 1.f, OUT_F32, 1);
  k_assemble<<<cdiv((long long)MROWS * EMB, 256), 256, 0, stream>>>(
      tok_f, cls_tok, pos_emb, h_f);

  // ---- transformer layers
  for (int i = 0; i < DEP; ++i) {
    const unsigned short* qw = qkvw_b + (size_t)i * EMB * 3 * EMB;
    const unsigned short* pw = projw_b + (size_t)i * EMB * EMB;
    const unsigned short* m1 = w1_b + (size_t)i * EMB * MLPD;
    const unsigned short* m2 = w2_b + (size_t)i * MLPD * EMB;

    // LN1 -> bf16
    k_layernorm<<<cdiv(MROWS, 8), 256, 0, stream>>>(
        h_f, ln1_w + i * EMB, ln1_b + i * EMB, yb, MROWS);

    // QKV: (6304x768)(768x2304) -> bf16
    gemm(yb, 0, qw, 0, qkv_b + i * 3 * EMB, qkvb, 1, 0, 0,
         MROWS, 3 * EMB, EMB, 3 * EMB, 1.f, OUT_BF16, 1);

    // reorder to Q / K^T / V (padded)
    k_reorder<<<cdiv((long long)BH * SPAD * HDIM, 256), 256, 0, stream>>>(
        qkvb, q_c, kT, v_c);

    // scores = (Q K^T) / 8 : batched Z=384, M=197, N=224, K=64 -> f32
    gemm(q_c, (long long)SEQ * HDIM, kT, (long long)HDIM * SPAD, nullptr,
         scores, 1, (long long)SEQ * SPAD, 0,
         SEQ, SPAD, HDIM, SPAD, 0.125f, OUT_F32, BH);

    // softmax rows -> bf16 probs (padding zeroed)
    k_softmax<<<cdiv((long long)BH * SEQ, 8), 256, 0, stream>>>(
        scores, probs, BH * SEQ);

    // O = P V : batched, M=197, N=64, K=224 -> o_b[(b*S+s)*E + h*64 + d]
    gemm(probs, (long long)SEQ * SPAD, v_c, (long long)SPAD * HDIM, nullptr,
         o_b, NH, (long long)SEQ * EMB, (long long)HDIM,
         SEQ, HDIM, SPAD, EMB, 1.f, OUT_BF16, BH);

    // h += O proj_w + proj_b
    gemm(o_b, 0, pw, 0, proj_b + i * EMB, h_f, 1, 0, 0,
         MROWS, EMB, EMB, EMB, 1.f, OUT_ADD_F32, 1);

    // LN2 -> bf16
    k_layernorm<<<cdiv(MROWS, 8), 256, 0, stream>>>(
        h_f, ln2_w + i * EMB, ln2_b + i * EMB, yb, MROWS);

    // MLP up + exact GELU -> bf16
    gemm(yb, 0, m1, 0, mlp_b1 + i * MLPD, mb, 1, 0, 0,
         MROWS, MLPD, EMB, MLPD, 1.f, OUT_GELU_BF16, 1);

    // h += MLP down
    gemm(mb, 0, m2, 0, mlp_b2 + i * EMB, h_f, 1, 0, 0,
         MROWS, EMB, MLPD, EMB, 1.f, OUT_ADD_F32, 1);
  }

  // ---- final LN (cls rows) + classification head
  k_head<<<BATCH, 32, 0, stream>>>(h_f, lnf_w, lnf_b, head_w, head_b, out);
}
